// RadMPNN_14937896255738
// MI455X (gfx1250) — compile-verified, matrix operand order checked by reading, not run on hardware
//
#include <hip/hip_runtime.h>

typedef __attribute__((ext_vector_type(16))) _Float16 v16h;
typedef __attribute__((ext_vector_type(8)))  float    v8f;

#define NN   50000
#define NE   800000
#define DD   64
#define WPB  4        // waves per block
#define TBLK 128      // 4 * wave32

__device__ __forceinline__ float gelu_f(float x) {
  return 0.5f * x * (1.0f + erff(x * 0.70710678118654752440f));
}

__device__ __forceinline__ unsigned int pack2h(float a, float b) {
  _Float16 ha = (_Float16)a, hb = (_Float16)b;
  unsigned short ua, ub;
  __builtin_memcpy(&ua, &ha, 2);
  __builtin_memcpy(&ub, &hb, 2);
  return (unsigned int)ua | ((unsigned int)ub << 16);
}

// Intra-wave LDS producer -> consumer ordering (tiles are wave-private;
// lanes run in lockstep, only DS completion matters -> no block barrier).
__device__ __forceinline__ void lds_fence() {
  asm volatile("s_wait_dscnt 0x0" ::: "memory");
}

// ---------------------------------------------------------------------------
// Weight staging: convert f32 [K][64] global weights into fragment-major f16
// LDS layout so each B fragment is 16 contiguous halves per lane (2x b128):
//   Ws[((kc*4 + t)*32 + lane)*16 + i]
// maps to W[k][n] with  n = t*16 + (lane&15),
//   k = kc*32 + ((lane>>4)<<3) + (i<8 ? i : 8+i)   (CDNA5 16-bit B layout).
// Rows k >= Krows are zero padding.
// ---------------------------------------------------------------------------
template<int KCH>
__device__ __forceinline__ void stage_w_swizzled(const float* __restrict__ Wg,
                                                 int Krows,
                                                 _Float16* __restrict__ Ws,
                                                 int tid)
{
  const int total = KCH * 2048;   // KCH * 4 tiles * 32 lanes * 16 halves
  for (int sidx = tid; sidx < total; sidx += TBLK) {
    const int i   = sidx & 15;
    const int L   = (sidx >> 4) & 31;
    const int t   = (sidx >> 9) & 3;
    const int kc  = sidx >> 11;
    const int off = (L >> 4) << 3;
    const int k   = kc * 32 + off + ((i < 8) ? i : (8 + i));
    const int n   = t * 16 + (L & 15);
    const float w = (k < Krows) ? Wg[k * DD + n] : 0.0f;
    Ws[sidx] = (_Float16)w;
  }
}

// ---------------------------------------------------------------------------
// One wave: D(16xDD) = A(16xK) * W(KxDD) + bias with V_WMMA_F32_16X16X32_F16.
// A is [16][KPAD] halves in LDS (row-contiguous), W is pre-swizzled (above).
// Fragment layouts per CDNA5 ISA 7.12.2. Result staged f32 to Dt[16*DD].
// ---------------------------------------------------------------------------
template<int KCH, int KPAD>
__device__ __forceinline__ void wave_gemm(const _Float16* __restrict__ At,
                                          const _Float16* __restrict__ Ws,
                                          const float* __restrict__ bias,
                                          float* __restrict__ Dt, int lane)
{
  v8f acc[4] = {};
  const int m   = lane & 15;
  const int off = (lane >> 4) << 3;   // 0 or 8
  #pragma unroll
  for (int kc = 0; kc < KCH; ++kc) {
    union { uint4 q[2]; v16h v; } au;
    const _Float16* ap = At + m * KPAD + kc * 32 + off;
    au.q[0] = *(const uint4*)(ap);        // K offs {0..7}+off
    au.q[1] = *(const uint4*)(ap + 16);   // K offs {16..23}+off
    #pragma unroll
    for (int t = 0; t < 4; ++t) {
      union { uint4 q[2]; v16h v; } bu;
      const _Float16* bp = Ws + (((kc * 4 + t) * 32) + lane) * 16;
      bu.q[0] = *(const uint4*)(bp);
      bu.q[1] = *(const uint4*)(bp + 8);
      acc[t] = __builtin_amdgcn_wmma_f32_16x16x32_f16(
          false, au.v, false, bu.v, (short)0, acc[t], false, false);
    }
  }
  const int mb = (lane >> 4) << 3;
  #pragma unroll
  for (int t = 0; t < 4; ++t) {
    const int n = t * 16 + m;
    const float bn = bias[n];
    #pragma unroll
    for (int r = 0; r < 8; ++r)
      Dt[(r + mb) * DD + n] = acc[t][r] + bn;
  }
}

// ---------------------------------------------------------------------------
// Kernel 1: fused edge featurization (RBF + SH-l2) + edge GEMM + GeLU + LN.
// Also accumulates in-degree counts. Emits edge_attr as f16 (WMMA A reuse).
// ---------------------------------------------------------------------------
__global__ void __launch_bounds__(TBLK)
k_edge(const float* __restrict__ pos, const int* __restrict__ eidx,
       const float* __restrict__ eW, const float* __restrict__ eb,
       const float* __restrict__ eg, const float* __restrict__ ebeta,
       _Float16* __restrict__ eattr, float* __restrict__ counts)
{
  constexpr int EK  = 96;   // 80 features padded to 3 K-chunks of 32
  constexpr int KCH = 3;
  __shared__ __align__(16) _Float16 sW[KCH * 2048];
  __shared__ __align__(16) _Float16 sA[WPB][16 * EK];
  __shared__ __align__(16) float    sD[WPB][16 * DD];

  stage_w_swizzled<KCH>(eW, 80, sW, threadIdx.x);
  __syncthreads();

  const int wave = threadIdx.x >> 5;
  const int lane = threadIdx.x & 31;
  const long e0 = ((long)blockIdx.x * WPB + wave) * 16;
  _Float16* At = sA[wave];
  float*    Dt = sD[wave];

  if (lane < 16) {
    const long e = e0 + lane;
    const int nr = eidx[e];
    const int nc = eidx[NE + e];
    const float rx = pos[nr * 3 + 0] - pos[nc * 3 + 0];
    const float ry = pos[nr * 3 + 1] - pos[nc * 3 + 1];
    const float rz = pos[nr * 3 + 2] - pos[nc * 3 + 2];
    const float dist = sqrtf(rx * rx + ry * ry + rz * rz);
    const float incut = (dist < 10.0f) ? 1.0f : 0.0f;
    const float fcut  = 0.5f * (__cosf(dist * 0.31415926535897932f) + 1.0f) * incut;
    const float sc    = fcut / (dist + 1e-8f);
    float rbf[16];
    #pragma unroll
    for (int k = 0; k < 16; ++k)
      rbf[k] = __sinf((float)(k + 1) * 0.31415926535897932f * dist) * sc;
    const float invn = 1.0f / (dist + 1e-10f);
    const float x = rx * invn, y = ry * invn, z = rz * invn;
    float sph[9];
    sph[0] = 0.2820947917738781f;
    sph[1] = 0.4886025119029199f * y;
    sph[2] = 0.4886025119029199f * z;
    sph[3] = 0.4886025119029199f * x;
    sph[4] = 0.5462742152960396f * x * y;
    sph[5] = 0.5462742152960396f * y * z;
    sph[6] = 0.6307831305050401f * 0.5f * (3.0f * z * z - 1.0f);
    sph[7] = 0.5462742152960396f * x * z;
    sph[8] = 0.5462742152960396f * 0.5f * (x * x - y * y);
    #pragma unroll
    for (int s = 0; s < 9; ++s) sph[s] *= incut;

    unsigned int* du = (unsigned int*)(At + lane * EK);
    #pragma unroll
    for (int j = 0; j < EK / 2; ++j) {
      const int i0 = 2 * j, i1 = 2 * j + 1;
      const float f0 = (i0 < 16) ? rbf[i0] * sph[0]
                     : (i0 < 80 ? rbf[(i0 - 16) >> 3] * sph[1 + ((i0 - 16) & 7)] : 0.0f);
      const float f1 = (i1 < 16) ? rbf[i1] * sph[0]
                     : (i1 < 80 ? rbf[(i1 - 16) >> 3] * sph[1 + ((i1 - 16) & 7)] : 0.0f);
      du[j] = pack2h(f0, f1);
    }
    atomicAdd(&counts[nc], 1.0f);
  }
  lds_fence();

  wave_gemm<KCH, EK>(At, sW, eb, Dt, lane);
  lds_fence();

  // GeLU + LayerNorm: 2 lanes per row (wave32), cross-half via shfl_xor(16).
  const int m  = lane & 15;
  const int c0 = (lane >> 4) * 32;
  const float4* dp = (const float4*)(Dt + m * DD + c0);
  float v[32], s = 0.f, s2 = 0.f;
  #pragma unroll
  for (int jq = 0; jq < 8; ++jq) {
    const float4 q = dp[jq];
    v[4 * jq + 0] = gelu_f(q.x); v[4 * jq + 1] = gelu_f(q.y);
    v[4 * jq + 2] = gelu_f(q.z); v[4 * jq + 3] = gelu_f(q.w);
  }
  #pragma unroll
  for (int j = 0; j < 32; ++j) { s += v[j]; s2 += v[j] * v[j]; }
  s  += __shfl_xor(s, 16, 32);
  s2 += __shfl_xor(s2, 16, 32);
  const float mean = s * (1.0f / 64.0f);
  const float inv  = rsqrtf(s2 * (1.0f / 64.0f) - mean * mean + 1e-5f);
  unsigned int* op = (unsigned int*)(eattr + (e0 + m) * DD + c0);
  #pragma unroll
  for (int j = 0; j < 16; ++j) {
    const float a0 = (v[2 * j]     - mean) * inv * eg[c0 + 2 * j]     + ebeta[c0 + 2 * j];
    const float a1 = (v[2 * j + 1] - mean) * inv * eg[c0 + 2 * j + 1] + ebeta[c0 + 2 * j + 1];
    op[j] = pack2h(a0, a1);
  }
}

// ---------------------------------------------------------------------------
// Kernel 2: h (f32) -> h16 (f16); only needed once before step 0
// (k_upd maintains h16 afterwards).
// ---------------------------------------------------------------------------
__global__ void k_h2h(const float* __restrict__ h, _Float16* __restrict__ h16, int n)
{
  const int i = blockIdx.x * blockDim.x + threadIdx.x;
  if (i < n) h16[i] = (_Float16)h[i];
}

// ---------------------------------------------------------------------------
// Kernel 3: message GEMM [h[row]|h[col]|eattr](16x192) x W(192x64),
// GeLU + LN, float-atomic scatter into agg[col].
// ---------------------------------------------------------------------------
__global__ void __launch_bounds__(TBLK)
k_msg(const _Float16* __restrict__ h16, const _Float16* __restrict__ eattr,
      const int* __restrict__ eidx,
      const float* __restrict__ Wg, const float* __restrict__ bg,
      const float* __restrict__ gg, const float* __restrict__ betag,
      float* __restrict__ agg)
{
  constexpr int MK  = 192;
  constexpr int KCH = 6;
  __shared__ __align__(16) _Float16 sW[KCH * 2048];
  __shared__ __align__(16) _Float16 sA[WPB][16 * MK];
  __shared__ __align__(16) float    sD[WPB][16 * DD];

  stage_w_swizzled<KCH>(Wg, MK, sW, threadIdx.x);
  __syncthreads();

  const int wave = threadIdx.x >> 5;
  const int lane = threadIdx.x & 31;
  const long e0 = ((long)blockIdx.x * WPB + wave) * 16;
  _Float16* At = sA[wave];
  float*    Dt = sD[wave];

  const int m    = lane & 15;
  const int part = lane >> 4;          // which 32-column half this lane stages
  const long e = e0 + m;
  const int nr = eidx[e];
  const int nc = eidx[NE + e];

  // Gather A tile: 3 contiguous 64-half segments per row; b128 loads.
  const uint4* g0 = (const uint4*)(h16 + (long)nr * DD + part * 32);
  const uint4* g1 = (const uint4*)(h16 + (long)nc * DD + part * 32);
  const uint4* g2 = (const uint4*)(eattr + e * DD + part * 32);
  uint4* d0 = (uint4*)(At + m * MK +   0 + part * 32);
  uint4* d1 = (uint4*)(At + m * MK +  64 + part * 32);
  uint4* d2 = (uint4*)(At + m * MK + 128 + part * 32);
  #pragma unroll
  for (int j = 0; j < 4; ++j) { d0[j] = g0[j]; d1[j] = g1[j]; d2[j] = g2[j]; }
  lds_fence();

  wave_gemm<KCH, MK>(At, sW, bg, Dt, lane);
  lds_fence();

  const int c0 = part * 32;
  const float4* dp = (const float4*)(Dt + m * DD + c0);
  float v[32], s = 0.f, s2 = 0.f;
  #pragma unroll
  for (int jq = 0; jq < 8; ++jq) {
    const float4 q = dp[jq];
    v[4 * jq + 0] = gelu_f(q.x); v[4 * jq + 1] = gelu_f(q.y);
    v[4 * jq + 2] = gelu_f(q.z); v[4 * jq + 3] = gelu_f(q.w);
  }
  #pragma unroll
  for (int j = 0; j < 32; ++j) { s += v[j]; s2 += v[j] * v[j]; }
  s  += __shfl_xor(s, 16, 32);
  s2 += __shfl_xor(s2, 16, 32);
  const float mean = s * (1.0f / 64.0f);
  const float inv  = rsqrtf(s2 * (1.0f / 64.0f) - mean * mean + 1e-5f);
  float* ap = agg + (long)nc * DD + c0;
  #pragma unroll
  for (int j = 0; j < 32; ++j)
    atomicAdd(&ap[j], (v[j] - mean) * inv * gg[c0 + j] + betag[c0 + j]);
}

// ---------------------------------------------------------------------------
// Kernel 4: update GEMM [h|agg/denom](16x128) x W(128x64), LN, h += upd.
// Also refreshes h16 for the next step's gathers.
// ---------------------------------------------------------------------------
__global__ void __launch_bounds__(TBLK)
k_upd(float* __restrict__ h, _Float16* __restrict__ h16,
      const float* __restrict__ agg, const float* __restrict__ counts,
      const float* __restrict__ Wg, const float* __restrict__ bg,
      const float* __restrict__ gg, const float* __restrict__ betag)
{
  constexpr int UK    = 128;
  constexpr int KCH   = 4;
  constexpr int NTILE = NN / 16;  // 3125 (exact)
  __shared__ __align__(16) _Float16 sW[KCH * 2048];
  __shared__ __align__(16) _Float16 sA[WPB][16 * UK];
  __shared__ __align__(16) float    sD[WPB][16 * DD];

  stage_w_swizzled<KCH>(Wg, UK, sW, threadIdx.x);
  __syncthreads();

  const int wave = threadIdx.x >> 5;
  const int lane = threadIdx.x & 31;
  long tile = (long)blockIdx.x * WPB + wave;
  const bool valid = tile < NTILE;
  if (!valid) tile = 0;               // keep EXEC all-ones for WMMA; mask stores
  _Float16* At = sA[wave];
  float*    Dt = sD[wave];

  const int m    = lane & 15;
  const int part = lane >> 4;
  const long n = tile * 16 + m;
  const float rden = 1.0f / fmaxf(counts[n], 1.0f);

  const float* hp = h   + n * DD + part * 32;
  const float* ap = agg + n * DD + part * 32;
  unsigned int* dh = (unsigned int*)(At + m * UK +  0 + part * 32);
  unsigned int* da = (unsigned int*)(At + m * UK + 64 + part * 32);
  float hold[32];
  #pragma unroll
  for (int j = 0; j < 16; ++j) {
    hold[2 * j]     = hp[2 * j];
    hold[2 * j + 1] = hp[2 * j + 1];
    dh[j] = pack2h(hold[2 * j], hold[2 * j + 1]);
    da[j] = pack2h(ap[2 * j] * rden, ap[2 * j + 1] * rden);
  }
  lds_fence();

  wave_gemm<KCH, UK>(At, sW, bg, Dt, lane);
  lds_fence();

  const int c0 = part * 32;
  const float4* dp = (const float4*)(Dt + m * DD + c0);
  float v[32], s = 0.f, s2 = 0.f;
  #pragma unroll
  for (int jq = 0; jq < 8; ++jq) {
    const float4 q = dp[jq];
    v[4 * jq + 0] = q.x; v[4 * jq + 1] = q.y;
    v[4 * jq + 2] = q.z; v[4 * jq + 3] = q.w;
  }
  #pragma unroll
  for (int j = 0; j < 32; ++j) { s += v[j]; s2 += v[j] * v[j]; }
  s  += __shfl_xor(s, 16, 32);
  s2 += __shfl_xor(s2, 16, 32);
  const float mean = s * (1.0f / 64.0f);
  const float inv  = rsqrtf(s2 * (1.0f / 64.0f) - mean * mean + 1e-5f);
  if (valid) {
    float* op = h + n * DD + c0;
    unsigned int* oh = (unsigned int*)(h16 + n * DD + c0);
    #pragma unroll
    for (int j = 0; j < 16; ++j) {
      const float n0 = hold[2 * j]     + (v[2 * j]     - mean) * inv * gg[c0 + 2 * j]     + betag[c0 + 2 * j];
      const float n1 = hold[2 * j + 1] + (v[2 * j + 1] - mean) * inv * gg[c0 + 2 * j + 1] + betag[c0 + 2 * j + 1];
      op[2 * j]     = n0;
      op[2 * j + 1] = n1;
      oh[j] = pack2h(n0, n1);
    }
  }
}

// ---------------------------------------------------------------------------
extern "C" void kernel_launch(void* const* d_in, const int* in_sizes, int n_in,
                              void* d_out, int out_size, void* d_ws, size_t ws_size,
                              hipStream_t stream)
{
  (void)in_sizes; (void)n_in; (void)out_size; (void)ws_size;
  const float* sf    = (const float*)d_in[0];   // scalar_features [N,64]
  const float* pos   = (const float*)d_in[1];   // cartesian_pos   [N,3]
  const int*   eidx  = (const int*)  d_in[2];   // edge_index      [2,E]
  const float* eW    = (const float*)d_in[3];   // edge_W  [80,64]
  const float* eb    = (const float*)d_in[4];
  const float* eg    = (const float*)d_in[5];
  const float* ebeta = (const float*)d_in[6];
  const float* msgW  = (const float*)d_in[7];   // [4,192,64]
  const float* msgb  = (const float*)d_in[8];
  const float* msgg  = (const float*)d_in[9];
  const float* msgbe = (const float*)d_in[10];
  const float* updW  = (const float*)d_in[11];  // [4,128,64]
  const float* updb  = (const float*)d_in[12];
  const float* updg  = (const float*)d_in[13];
  const float* updbe = (const float*)d_in[14];

  float* h = (float*)d_out;                     // h lives in d_out, updated in place

  // Workspace layout (~122 MB): counts | agg | h16 | edge_attr(f16)
  char* ws = (char*)d_ws;
  size_t off = 0;
  float* counts = (float*)(ws + off);  off += (((size_t)NN * 4) + 255) & ~(size_t)255;
  float* agg    = (float*)(ws + off);  off += (size_t)NN * DD * 4;
  _Float16* h16   = (_Float16*)(ws + off); off += (size_t)NN * DD * 2;
  _Float16* eattr = (_Float16*)(ws + off);

  // h = scalar_features; h16 = f16(h); counts = 0
  hipMemcpyAsync(h, sf, (size_t)NN * DD * sizeof(float),
                 hipMemcpyDeviceToDevice, stream);
  hipMemsetAsync(counts, 0, (size_t)NN * sizeof(float), stream);

  const int h2hBlocks = (NN * DD + 255) / 256;
  k_h2h<<<h2hBlocks, 256, 0, stream>>>(h, h16, NN * DD);

  const int edgeBlocks = NE / (16 * WPB);          // 12500
  k_edge<<<edgeBlocks, TBLK, 0, stream>>>(pos, eidx, eW, eb, eg, ebeta,
                                          eattr, counts);

  const int updBlocks = (NN / 16 + WPB - 1) / WPB; // 782

  for (int s = 0; s < 4; ++s) {
    hipMemsetAsync(agg, 0, (size_t)NN * DD * sizeof(float), stream);
    k_msg<<<edgeBlocks, TBLK, 0, stream>>>(h16, eattr, eidx,
                                           msgW + (size_t)s * 192 * DD,
                                           msgb + s * DD, msgg + s * DD,
                                           msgbe + s * DD, agg);
    k_upd<<<updBlocks, TBLK, 0, stream>>>(h, h16, agg, counts,
                                          updW + (size_t)s * 128 * DD,
                                          updb + s * DD, updg + s * DD,
                                          updbe + s * DD);
  }
}
